// TinySeq2Seq_52682068853381
// MI455X (gfx1250) — matrix operand order repaired and checked by reading, not compiled, not run on hardware
//
#include <hip/hip_runtime.h>
#include <hip/hip_bf16.h>
#include <math.h>

#define B_    8
#define L_    512
#define S_    513
#define D_    1024
#define H_    16
#define HD_   64
#define NL_   6
#define FF_   4096
#define MAXD_ 64
#define NBIAS_ 129
#define OUTL_ 8
#define NC_   10
#define M_    (B_*S_)      /* 4104 rows across all batches */
#define LN_EPS 1e-5f

typedef __bf16 bf16;
typedef __attribute__((ext_vector_type(16))) __bf16 v16bf;
typedef __attribute__((ext_vector_type(8)))  float  v8f;

__device__ __forceinline__ unsigned short f2bfu(float f) {
  unsigned u = __builtin_bit_cast(unsigned, f);
  return (unsigned short)((u + 0x7FFFu + ((u >> 16) & 1u)) >> 16);   // RNE
}
__device__ __forceinline__ bf16 f2bf(float f) {
  return __builtin_bit_cast(bf16, f2bfu(f));
}

// ---------------------------------------------------------------------------
// Embedding: tok + seq-pos + digit-pos, CLS prepended. Writes fp32 + bf16.
// ---------------------------------------------------------------------------
__launch_bounds__(256)
__global__ void embed_kernel(const int* __restrict__ src,
                             const float* __restrict__ tok_emb,
                             const float* __restrict__ seq_pos,
                             const float* __restrict__ dig_pos,
                             const float* __restrict__ cls_tok,
                             float* __restrict__ xF, bf16* __restrict__ xB) {
  __shared__ short dpos[L_];
  const int b = blockIdx.x;
  if (threadIdx.x == 0) {
    int cnt = 0;
    for (int j = L_ - 1; j >= 0; --j) {
      int t = src[b*L_ + j];
      if (t < NC_) { dpos[j] = (short)cnt; ++cnt; }
      else         { dpos[j] = -1; cnt = 0; }
    }
  }
  __syncthreads();
  for (int i = threadIdx.x; i < S_*D_; i += blockDim.x) {
    const int s = i / D_, d = i - s*D_;
    float v;
    if (s == 0) {
      v = cls_tok[d];
    } else {
      const int j = s - 1;
      const int t = src[b*L_ + j];
      v = tok_emb[t*D_ + d] + seq_pos[j*D_ + d];
      const int dp = dpos[j];
      if (dp >= 0) v += dig_pos[dp*D_ + d];
    }
    const size_t row = (size_t)(b*S_ + s);
    xF[row*D_ + d] = v;
    xB[row*D_ + d] = f2bf(v);
  }
}

// ---------------------------------------------------------------------------
// bf16 WMMA GEMM:  C[M,N] = A[M,K](bf16) @ W[K,N](fp32->bf16) + bias, opt GELU
// Block: 256 threads = 8 waves, tile 128x64, K-step 32, register-pipelined
// double-buffered LDS. Per K-step and wave:
//   (1) 10x ds_load_b128 -> A fragment + 4 B fragments (registers)
//   (2) next tile's global loads into registers (latency hidden under WMMAs)
//   (3) 4x v_wmma_f32_16x16x32_bf16 back-to-back (single DScnt wait)
//   (4) cvt + ds_store of staged registers into the other buffer
//   (5) one barrier
// ---------------------------------------------------------------------------
__launch_bounds__(256)
__global__ void gemm_bf16(const bf16* __restrict__ A, const float* __restrict__ W,
                          const float* __restrict__ bias,
                          float* __restrict__ outF, bf16* __restrict__ outB,
                          int Mrows, int N, int K, int act) {
  __shared__ __align__(16) bf16 As [2][128*32];   // 2 x 8 KB
  __shared__ __align__(16) bf16 BsT[2][64*40];    // 2 x 5 KB (transposed, stride 40)

  const int tid  = threadIdx.x;
  const int lane = tid & 31;
  const int w    = tid >> 5;     // wave 0..7 -> rows [w*16, w*16+16)
  const int half = lane >> 4;
  const int mrow = lane & 15;
  const int n0   = blockIdx.x * 64;
  const int m0   = blockIdx.y * 128;

  // A staging: thread -> (tile row, 16-elem K chunk); row clamped once
  // (garbage rows only feed outputs that are never stored).
  const int ar = tid >> 1;
  const int ac = (tid & 1) * 16;
  int arow_g = m0 + ar; if (arow_g >= Mrows) arow_g = Mrows - 1;
  const bf16* aptr = A + (size_t)arow_g * K + ac;

  // B staging: thread -> (K pair kp, column group cg of 4 cols).
  const int cg = tid & 15;
  const int kp = tid >> 4;
  const float* wptr = W + (size_t)(2*kp) * N + n0 + cg*4;
  const size_t wstep = (size_t)32 * N;

  v8f c0 = {}, c1 = {}, c2 = {}, c3 = {};
  union UV { uint4 q[2]; v16bf v; };

  const int KT = K >> 5;

  // prologue: stage tile 0 into buffer 0
  {
    const uint4* s = (const uint4*)(aptr);
    uint4 a0 = s[0], a1 = s[1];
    float4 r0 = *(const float4*)(wptr);
    float4 r1 = *(const float4*)(wptr + N);
    uint4* dst = (uint4*)(&As[0][ar*32 + ac]);
    dst[0] = a0; dst[1] = a1;
    unsigned q0 = (unsigned)f2bfu(r0.x) | ((unsigned)f2bfu(r1.x) << 16);
    unsigned q1 = (unsigned)f2bfu(r0.y) | ((unsigned)f2bfu(r1.y) << 16);
    unsigned q2 = (unsigned)f2bfu(r0.z) | ((unsigned)f2bfu(r1.z) << 16);
    unsigned q3 = (unsigned)f2bfu(r0.w) | ((unsigned)f2bfu(r1.w) << 16);
    bf16* base = &BsT[0][0];
    *(unsigned*)(base + (cg*4+0)*40 + 2*kp) = q0;
    *(unsigned*)(base + (cg*4+1)*40 + 2*kp) = q1;
    *(unsigned*)(base + (cg*4+2)*40 + 2*kp) = q2;
    *(unsigned*)(base + (cg*4+3)*40 + 2*kp) = q3;
  }
  __syncthreads();

  for (int kt = 0; kt < KT; ++kt) {
    const int cur = kt & 1;
    const bool more = (kt + 1 < KT);

    // (1) all fragments for this K-step -> registers
    UV ua, ub0, ub1, ub2, ub3;
    {
      const bf16* arw = &As[cur][(w*16 + mrow)*32];
      ua.q[0] = *(const uint4*)(arw + half*8);
      ua.q[1] = *(const uint4*)(arw + 16 + half*8);
      const bf16* bb = &BsT[cur][mrow*40 + half*8];
      ub0.q[0] = ((const uint4*)(bb        ))[0]; ub0.q[1] = ((const uint4*)(bb        ))[2];
      ub1.q[0] = ((const uint4*)(bb + 16*40))[0]; ub1.q[1] = ((const uint4*)(bb + 16*40))[2];
      ub2.q[0] = ((const uint4*)(bb + 32*40))[0]; ub2.q[1] = ((const uint4*)(bb + 32*40))[2];
      ub3.q[0] = ((const uint4*)(bb + 48*40))[0]; ub3.q[1] = ((const uint4*)(bb + 48*40))[2];
    }

    // (2) next tile's global loads -> registers (overlap with WMMAs)
    uint4 a0 = {}, a1 = {};
    float4 r0 = {}, r1 = {};
    if (more) {
      const bf16* ap = aptr + (size_t)(kt + 1) * 32;
      a0 = ((const uint4*)ap)[0];
      a1 = ((const uint4*)ap)[1];
      const float* p = wptr + (size_t)(kt + 1) * wstep;
      r0 = *(const float4*)(p);
      r1 = *(const float4*)(p + N);
      __builtin_prefetch(wptr + (size_t)(kt + 2) * wstep, 0, 1);
    }

    // (3) 4 WMMAs back-to-back
    c0 = __builtin_amdgcn_wmma_f32_16x16x32_bf16(false, ua.v, false, ub0.v, (short)0, c0, false, false);
    c1 = __builtin_amdgcn_wmma_f32_16x16x32_bf16(false, ua.v, false, ub1.v, (short)0, c1, false, false);
    c2 = __builtin_amdgcn_wmma_f32_16x16x32_bf16(false, ua.v, false, ub2.v, (short)0, c2, false, false);
    c3 = __builtin_amdgcn_wmma_f32_16x16x32_bf16(false, ua.v, false, ub3.v, (short)0, c3, false, false);

    // (4) store staged registers into the other buffer
    if (more) {
      const int nxt = cur ^ 1;
      uint4* dst = (uint4*)(&As[nxt][ar*32 + ac]);
      dst[0] = a0; dst[1] = a1;
      unsigned q0 = (unsigned)f2bfu(r0.x) | ((unsigned)f2bfu(r1.x) << 16);
      unsigned q1 = (unsigned)f2bfu(r0.y) | ((unsigned)f2bfu(r1.y) << 16);
      unsigned q2 = (unsigned)f2bfu(r0.z) | ((unsigned)f2bfu(r1.z) << 16);
      unsigned q3 = (unsigned)f2bfu(r0.w) | ((unsigned)f2bfu(r1.w) << 16);
      bf16* base = &BsT[nxt][0];
      *(unsigned*)(base + (cg*4+0)*40 + 2*kp) = q0;
      *(unsigned*)(base + (cg*4+1)*40 + 2*kp) = q1;
      *(unsigned*)(base + (cg*4+2)*40 + 2*kp) = q2;
      *(unsigned*)(base + (cg*4+3)*40 + 2*kp) = q3;
    }
    __syncthreads();   // (5) next iter reads other buffer, restages this one
  }

  // --- epilogue: bias (+ exact GELU), dual-precision stores ---
  v8f acc[4] = { c0, c1, c2, c3 };
  #pragma unroll
  for (int t = 0; t < 4; ++t) {
    const int gc = n0 + t*16 + mrow;
    const float bs = bias ? bias[gc] : 0.0f;
    #pragma unroll
    for (int r = 0; r < 8; ++r) {
      const int grow = m0 + w*16 + half*8 + r;  // C VGPR r: M=r (lanes 0-15) / 8+r (16-31)
      if (grow >= Mrows) continue;
      float v = acc[t][r] + bs;
      if (act == 1) v = 0.5f * v * (1.0f + erff(v * 0.70710678f));  // exact GELU
      if (outF) outF[(size_t)grow*N + gc] = v;
      if (outB) outB[(size_t)grow*N + gc] = f2bf(v);
    }
  }
}

// ---------------------------------------------------------------------------
// Attention: one block per (q, head, batch). Scores + rel-pos bias + mask,
// block softmax, AV. qkv layout [row, 3D]: Q | K | V. Writes bf16 context.
// ---------------------------------------------------------------------------
__launch_bounds__(64)
__global__ void attn_kernel(const float* __restrict__ qkv,
                            const float* __restrict__ rel_emb,   // layer base
                            const unsigned char* __restrict__ pad_mask,
                            bf16* __restrict__ ctxB) {
  __shared__ float sq[HD_];
  __shared__ float sc[S_];
  __shared__ float red[64];
  const int qs = blockIdx.x;
  const int h  = blockIdx.y;
  const int b  = blockIdx.z;
  const int t  = threadIdx.x;

  sq[t] = qkv[(size_t)(b*S_ + qs)*(3*D_) + h*HD_ + t];
  __syncthreads();

  const float scale = 0.125f;   // HD^-0.5
  for (int k = t; k < S_; k += 64) {
    const float* kp = qkv + (size_t)(b*S_ + k)*(3*D_) + D_ + h*HD_;
    float acc = 0.0f;
    #pragma unroll 8
    for (int d = 0; d < HD_; ++d) acc += sq[d] * kp[d];
    int rel = k - qs;
    rel = rel < -MAXD_ ? -MAXD_ : (rel > MAXD_ ? MAXD_ : rel);
    const float bias = rel_emb[(rel + MAXD_)*H_ + h];
    const bool masked = (k > 0) && (pad_mask[b*L_ + (k-1)] != 0);
    sc[k] = masked ? -1e30f : (acc*scale + bias);
  }
  __syncthreads();

  float mx = -1e30f;
  for (int k = t; k < S_; k += 64) mx = fmaxf(mx, sc[k]);
  red[t] = mx; __syncthreads();
  for (int s2 = 32; s2 > 0; s2 >>= 1) { if (t < s2) red[t] = fmaxf(red[t], red[t+s2]); __syncthreads(); }
  mx = red[0]; __syncthreads();

  float sm = 0.0f;
  for (int k = t; k < S_; k += 64) { float e = __expf(sc[k] - mx); sc[k] = e; sm += e; }
  red[t] = sm; __syncthreads();
  for (int s2 = 32; s2 > 0; s2 >>= 1) { if (t < s2) red[t] += red[t+s2]; __syncthreads(); }
  const float inv = 1.0f / red[0];
  __syncthreads();

  float acc = 0.0f;
  for (int k = 0; k < S_; ++k) {
    acc += sc[k] * qkv[(size_t)(b*S_ + k)*(3*D_) + 2*D_ + h*HD_ + t];
  }
  ctxB[(size_t)(b*S_ + qs)*D_ + h*HD_ + t] = f2bf(acc * inv);
}

// ---------------------------------------------------------------------------
// Fused residual + LayerNorm. One block per row; in-place safe.
// ---------------------------------------------------------------------------
__launch_bounds__(256)
__global__ void add_ln_kernel(const float* __restrict__ xin, const float* __restrict__ y,
                              const float* __restrict__ g, const float* __restrict__ bta,
                              float* __restrict__ xoutF, bf16* __restrict__ xoutB) {
  __shared__ float red[256];
  const int row = blockIdx.x;
  const int t = threadIdx.x;
  float v[4];
  float s = 0.0f;
  #pragma unroll
  for (int i = 0; i < 4; ++i) {
    const int d = t + i*256;
    v[i] = xin[(size_t)row*D_ + d] + y[(size_t)row*D_ + d];
    s += v[i];
  }
  red[t] = s; __syncthreads();
  for (int s2 = 128; s2 > 0; s2 >>= 1) { if (t < s2) red[t] += red[t+s2]; __syncthreads(); }
  const float mu = red[0] * (1.0f/D_);
  __syncthreads();
  s = 0.0f;
  #pragma unroll
  for (int i = 0; i < 4; ++i) { const float d0 = v[i] - mu; s += d0*d0; }
  red[t] = s; __syncthreads();
  for (int s2 = 128; s2 > 0; s2 >>= 1) { if (t < s2) red[t] += red[t+s2]; __syncthreads(); }
  const float rstd = rsqrtf(red[0]*(1.0f/D_) + LN_EPS);
  #pragma unroll
  for (int i = 0; i < 4; ++i) {
    const int d = t + i*256;
    const float o = (v[i] - mu)*rstd*g[d] + bta[d];
    xoutF[(size_t)row*D_ + d] = o;
    xoutB[(size_t)row*D_ + d] = f2bf(o);
  }
}

// ---------------------------------------------------------------------------
// Final norm on CLS row + classifier head (1024 x 80). One block per batch.
// ---------------------------------------------------------------------------
__launch_bounds__(128)
__global__ void head_kernel(const float* __restrict__ xF,
                            const float* __restrict__ g, const float* __restrict__ bta,
                            const float* __restrict__ cls_w, const float* __restrict__ cls_b,
                            float* __restrict__ out) {
  __shared__ float xn[D_];
  __shared__ float red[128];
  const int b = blockIdx.x, t = threadIdx.x;
  const float* xr = xF + (size_t)(b*S_)*D_;   // CLS row
  float s = 0.0f;
  for (int d = t; d < D_; d += 128) s += xr[d];
  red[t] = s; __syncthreads();
  for (int s2 = 64; s2 > 0; s2 >>= 1) { if (t < s2) red[t] += red[t+s2]; __syncthreads(); }
  const float mu = red[0] * (1.0f/D_);
  __syncthreads();
  s = 0.0f;
  for (int d = t; d < D_; d += 128) { const float dd = xr[d] - mu; s += dd*dd; }
  red[t] = s; __syncthreads();
  for (int s2 = 64; s2 > 0; s2 >>= 1) { if (t < s2) red[t] += red[t+s2]; __syncthreads(); }
  const float rstd = rsqrtf(red[0]*(1.0f/D_) + LN_EPS);
  for (int d = t; d < D_; d += 128) xn[d] = (xr[d] - mu)*rstd*g[d] + bta[d];
  __syncthreads();
  for (int o = t; o < OUTL_*NC_; o += 128) {
    float acc = cls_b[o];
    for (int d = 0; d < D_; ++d) acc += xn[d] * cls_w[d*(OUTL_*NC_) + o];
    out[b*OUTL_*NC_ + o] = acc;
  }
}

// ---------------------------------------------------------------------------
extern "C" void kernel_launch(void* const* d_in, const int* in_sizes, int n_in,
                              void* d_out, int out_size, void* d_ws, size_t ws_size,
                              hipStream_t stream) {
  const int*            src      = (const int*)d_in[0];
  const unsigned char*  pad      = (const unsigned char*)d_in[1];
  const float*          tok_emb  = (const float*)d_in[2];
  const float*          seq_pos  = (const float*)d_in[3];
  const float*          dig_pos  = (const float*)d_in[4];
  const float*          cls_tok  = (const float*)d_in[5];
  const float*          qkv_w    = (const float*)d_in[6];
  const float*          qkv_b    = (const float*)d_in[7];
  const float*          out_w    = (const float*)d_in[8];
  const float*          out_b    = (const float*)d_in[9];
  const float*          rel_emb  = (const float*)d_in[10];
  const float*          ln1_g    = (const float*)d_in[11];
  const float*          ln1_b    = (const float*)d_in[12];
  const float*          lin1_w   = (const float*)d_in[13];
  const float*          lin1_b   = (const float*)d_in[14];
  const float*          lin2_w   = (const float*)d_in[15];
  const float*          lin2_b   = (const float*)d_in[16];
  const float*          ln2_g    = (const float*)d_in[17];
  const float*          ln2_b    = (const float*)d_in[18];
  const float*          fn_g     = (const float*)d_in[19];
  const float*          fn_b     = (const float*)d_in[20];
  const float*          cls_w    = (const float*)d_in[21];
  const float*          cls_b    = (const float*)d_in[22];
  float* out = (float*)d_out;
  (void)in_sizes; (void)n_in; (void)out_size; (void)ws_size;

  // workspace carve-up (~135 MB)
  char* ws = (char*)d_ws;
  size_t off = 0;
  auto alloc = [&](size_t bytes) -> char* {
    char* p = ws + off;
    off += (bytes + 255) & ~(size_t)255;
    return p;
  };
  float* xF   = (float*)alloc((size_t)M_*D_*4);
  bf16*  xB   = (bf16*) alloc((size_t)M_*D_*2);
  float* qkvF = (float*)alloc((size_t)M_*3*D_*4);
  bf16*  ctxB = (bf16*) alloc((size_t)M_*D_*2);
  float* tmpF = (float*)alloc((size_t)M_*D_*4);
  bf16*  ff1B = (bf16*) alloc((size_t)M_*FF_*2);

  embed_kernel<<<dim3(B_), dim3(256), 0, stream>>>(src, tok_emb, seq_pos, dig_pos,
                                                   cls_tok, xF, xB);

  const int MT = (M_ + 127) / 128;  // 33 row-tiles of 128
  for (int l = 0; l < NL_; ++l) {
    // QKV projection: [M,1024] @ [1024,3072]
    gemm_bf16<<<dim3(3*D_/64, MT), dim3(256), 0, stream>>>(
        xB, qkv_w + (size_t)l*D_*3*D_, qkv_b + (size_t)l*3*D_,
        qkvF, nullptr, M_, 3*D_, D_, 0);
    // attention per (q, head, batch)
    attn_kernel<<<dim3(S_, H_, B_), dim3(64), 0, stream>>>(
        qkvF, rel_emb + (size_t)l*NBIAS_*H_, pad, ctxB);
    // output projection: [M,1024] @ [1024,1024]
    gemm_bf16<<<dim3(D_/64, MT), dim3(256), 0, stream>>>(
        ctxB, out_w + (size_t)l*D_*D_, out_b + (size_t)l*D_,
        tmpF, nullptr, M_, D_, D_, 0);
    add_ln_kernel<<<dim3(M_), dim3(256), 0, stream>>>(
        xF, tmpF, ln1_g + (size_t)l*D_, ln1_b + (size_t)l*D_, xF, xB);
    // FFN up + GELU: [M,1024] @ [1024,4096]
    gemm_bf16<<<dim3(FF_/64, MT), dim3(256), 0, stream>>>(
        xB, lin1_w + (size_t)l*D_*FF_, lin1_b + (size_t)l*FF_,
        nullptr, ff1B, M_, FF_, D_, 1);
    // FFN down: [M,4096] @ [4096,1024]
    gemm_bf16<<<dim3(D_/64, MT), dim3(256), 0, stream>>>(
        ff1B, lin2_w + (size_t)l*FF_*D_, lin2_b + (size_t)l*D_,
        tmpF, nullptr, M_, D_, FF_, 0);
    add_ln_kernel<<<dim3(M_), dim3(256), 0, stream>>>(
        xF, tmpF, ln2_g + (size_t)l*D_, ln2_b + (size_t)l*D_, xF, xB);
  }
  head_kernel<<<dim3(B_), dim3(128), 0, stream>>>(xF, fn_g, fn_b, cls_w, cls_b, out);
}
